// DTMLModel_17403207483544
// MI455X (gfx1250) — compile-verified, matrix operand order checked by reading, not run on hardware
//
#include <hip/hip_runtime.h>
#include <math.h>

typedef __attribute__((ext_vector_type(16))) _Float16 v16h;
typedef __attribute__((ext_vector_type(8)))  float    v8f;

static constexpr int BATCH = 512;
static constexpr int SEQ   = 256;
static constexpr int DIN   = 128;
static constexpr int HID   = 256;
static constexpr int GATE4 = 1024;   // 4*HID
static constexpr int NSTK  = 32;
static constexpr int DH    = 32;     // HID / NHEADS

// ---------------------------------------------------------------- helpers ----
__device__ __forceinline__ v8f zero8() {
  v8f z;
#pragma unroll
  for (int i = 0; i < 8; ++i) z[i] = 0.f;
  return z;
}

__device__ __forceinline__ v8f wmma16(v16h a, v16h b, v8f c) {
  return __builtin_amdgcn_wmma_f32_16x16x32_f16(
      /*neg_a=*/false, a, /*neg_b=*/false, b,
      /*c_mod=*/(short)0, c, /*reuse_a=*/false, /*reuse_b=*/false);
}

// Fragment with lane&15 selecting row r0+(lane&15), 32 consecutive K values per
// the CDNA5 16-bit A/B VGPR layout (lanes 0-15: K 0-7 & 16-23; lanes 16-31: +8).
__device__ __forceinline__ v16h load_frag_h(const _Float16* __restrict__ base,
                                            int ld, int r0, int c0) {
  const int lane = threadIdx.x & 31;
  const int hi   = lane >> 4;
  const _Float16* p = base + (size_t)(r0 + (lane & 15)) * ld + c0 + hi * 8;
  v16h f;
#pragma unroll
  for (int v = 0; v < 8; ++v) {
    int k = (v < 4) ? (2 * v) : (16 + 2 * (v - 4));
    f[2 * v]     = p[k];
    f[2 * v + 1] = p[k + 1];
  }
  return f;
}

__device__ __forceinline__ v16h load_frag_f32(const float* __restrict__ base,
                                              int ld, int r0, int c0) {
  const int lane = threadIdx.x & 31;
  const int hi   = lane >> 4;
  const float* p = base + (size_t)(r0 + (lane & 15)) * ld + c0 + hi * 8;
  v16h f;
#pragma unroll
  for (int v = 0; v < 8; ++v) {
    int k = (v < 4) ? (2 * v) : (16 + 2 * (v - 4));
    f[2 * v]     = (_Float16)p[k];
    f[2 * v + 1] = (_Float16)p[k + 1];
  }
  return f;
}

__device__ __forceinline__ float sigf(float x) { return 1.f / (1.f + __expf(-x)); }

// ----------------------------------------------------------- tiny utilities ----
__global__ void cvt_f32_f16(const float* __restrict__ s, _Float16* __restrict__ d, int n) {
  int i = blockIdx.x * 256 + threadIdx.x;
  if (i < n) d[i] = (_Float16)s[i];
}

__global__ void bias_comb(const float* __restrict__ a, const float* __restrict__ b,
                          float* __restrict__ o, int n) {
  int i = blockIdx.x * 256 + threadIdx.x;
  if (i < n) o[i] = a[i] + b[i];
}

// --------------------------------------------- y = tanh(x @ W.T + b) -> f16 ----
// grid = (B*S)/16 blocks, 256 threads (8 waves); wave w owns N-tile w (N=128).
__global__ __launch_bounds__(256) void transform_kernel(
    const float* __restrict__ x, const _Float16* __restrict__ wh,
    const float* __restrict__ bias, _Float16* __restrict__ out) {
  const int lane = threadIdx.x & 31, wv = threadIdx.x >> 5, hi = lane >> 4;
  const int m0 = blockIdx.x * 16;
  const int n0 = wv * 16;
  v8f acc = zero8();
#pragma unroll
  for (int kk = 0; kk < 4; ++kk) {
    v16h a = load_frag_f32(x, DIN, m0, kk * 32);
    v16h b = load_frag_h(wh, DIN, n0, kk * 32);
    acc = wmma16(a, b, acc);
  }
  const int n = n0 + (lane & 15);
  const float bv = bias[n];
#pragma unroll
  for (int v = 0; v < 8; ++v) {
    int m = m0 + v + 8 * hi;
    out[(size_t)m * DIN + n] = (_Float16)tanhf(acc[v] + bv);
  }
}

// ------------- gx = act @ wih.T (no bias), stored time-major -> f16 ----
// gx layout: [t][batch][GATE4] so the per-step 16-batch tile is contiguous.
__global__ __launch_bounds__(256) void gates_kernel(
    const _Float16* __restrict__ act, const _Float16* __restrict__ wih,
    _Float16* __restrict__ gx) {
  const int lane = threadIdx.x & 31, wv = threadIdx.x >> 5, hi = lane >> 4;
  const int m0 = blockIdx.x * 16;
  v16h a[4];
#pragma unroll
  for (int kk = 0; kk < 4; ++kk) a[kk] = load_frag_h(act, DIN, m0, kk * 32);
#pragma unroll
  for (int j = 0; j < 8; ++j) {
    int n0 = (wv * 8 + j) * 16;
    v8f acc = zero8();
#pragma unroll
    for (int kk = 0; kk < 4; ++kk) {
      v16h b = load_frag_h(wih, DIN, n0, kk * 32);
      acc = wmma16(a[kk], b, acc);
    }
    int n = n0 + (lane & 15);
#pragma unroll
    for (int v = 0; v < 8; ++v) {
      int m = m0 + v + 8 * hi;         // flat row = b*SEQ + t
      int bb = m >> 8;                 // m / SEQ
      int tt = m & (SEQ - 1);          // m % SEQ
      gx[((size_t)tt * BATCH + bb) * GATE4 + n] = (_Float16)acc[v];
    }
  }
}

// --------------------------------------------------- persistent LSTM + attn ----
// grid = BATCH/16 blocks; each block owns 16 batch rows; wave w owns hidden
// tiles {w, w+8} -> c state in registers; h in LDS f16; WMMA recurrence.
// The per-step gx tile (16 x 1024 f16 = 32 KB, contiguous) is staged into LDS
// with GLOBAL_LOAD_ASYNC_TO_LDS_B128, overlapped with the recurrent WMMAs.
__global__ __launch_bounds__(256) void lstm_kernel(
    const _Float16* __restrict__ gx, const _Float16* __restrict__ whh,
    const float* __restrict__ bias4, _Float16* __restrict__ hs,
    float* __restrict__ ctx) {
  __shared__ _Float16 hbuf[16 * HID];
  __shared__ float sbuf[16 * SEQ];
  __shared__ _Float16 gxbuf[16 * GATE4];
  const int tid = threadIdx.x, lane = tid & 31, wv = tid >> 5, hi = lane >> 4;
  const int b0 = blockIdx.x * 16;
  const unsigned gx_lds = (unsigned)(size_t)(&gxbuf[0]);  // LDS byte offset

  for (int i = tid; i < 16 * HID; i += 256) hbuf[i] = (_Float16)0.f;
  v8f cst[2];
  cst[0] = zero8(); cst[1] = zero8();

  for (int t = 0; t < SEQ; ++t) {
    __syncthreads();   // prev h writes + prev gxbuf reads complete

    // Issue async copy of this step's gx tile (2048 x 16B chunks, 8/thread).
    {
      const char* gbase = (const char*)(gx + ((size_t)t * BATCH + b0) * GATE4);
#pragma unroll
      for (int i = 0; i < 8; ++i) {
        unsigned c = (unsigned)tid + (unsigned)i * 256u;
        unsigned loff = gx_lds + c * 16u;
        unsigned long long gaddr =
            (unsigned long long)(size_t)gbase + (unsigned long long)c * 16ull;
        asm volatile("global_load_async_to_lds_b128 %0, %1, off"
                     :: "v"(loff), "v"(gaddr) : "memory");
      }
    }

    v16h a[8];
#pragma unroll
    for (int kk = 0; kk < 8; ++kk) a[kk] = load_frag_h(hbuf, HID, 0, kk * 32);
    __syncthreads();   // all A loads done before h rewrite

    v8f acc[8];
#pragma unroll
    for (int j = 0; j < 8; ++j) {        // tile T = wv + 8j: gate = j>>1, slot = j&1
      int n0 = (wv + 8 * j) * 16;
      v8f s = zero8();
#pragma unroll
      for (int kk = 0; kk < 8; ++kk) {
        v16h b = load_frag_h(whh, HID, n0, kk * 32);
        s = wmma16(a[kk], b, s);
      }
      acc[j] = s;
    }

    asm volatile("s_wait_asynccnt 0x0" ::: "memory");
    __syncthreads();   // every wave's async chunks visible to all waves

#pragma unroll
    for (int j = 0; j < 8; ++j) {
      int n = (wv + 8 * j) * 16 + (lane & 15);
      float bv = bias4[n];
      v8f s = acc[j];
#pragma unroll
      for (int v = 0; v < 8; ++v) {
        int m = v + 8 * hi;
        s[v] += bv + (float)gxbuf[m * GATE4 + n];
      }
      acc[j] = s;
    }

#pragma unroll
    for (int sl = 0; sl < 2; ++sl) {
      int col = (wv + 8 * sl) * 16 + (lane & 15);
#pragma unroll
      for (int v = 0; v < 8; ++v) {
        float iv = sigf(acc[0 + sl][v]);
        float fv = sigf(acc[2 + sl][v]);
        float gv = tanhf(acc[4 + sl][v]);
        float ov = sigf(acc[6 + sl][v]);
        float cc = fv * cst[sl][v] + iv * gv;
        cst[sl][v] = cc;
        float hv = ov * tanhf(cc);
        int m = v + 8 * hi;
        hbuf[m * HID + col] = (_Float16)hv;
        hs[((size_t)(b0 + m) * SEQ + t) * HID + col] = (_Float16)hv;
      }
    }
  }
  __syncthreads();

  // dot attention vs last h: score -> (score/sum) weighted context
  for (int r = wv * 2; r < wv * 2 + 2; ++r) {
    const int d0 = lane * 8;
    float hl[8];
#pragma unroll
    for (int i = 0; i < 8; ++i) hl[i] = (float)hbuf[r * HID + d0 + i];
    for (int t = 0; t < SEQ; ++t) {
      const _Float16* hp = hs + ((size_t)(b0 + r) * SEQ + t) * HID + d0;
      float s = 0.f;
#pragma unroll
      for (int i = 0; i < 8; ++i) s += hl[i] * (float)hp[i];
#pragma unroll
      for (int off = 16; off >= 1; off >>= 1) s += __shfl_xor(s, off, 32);
      if (lane == 0) sbuf[r * SEQ + t] = s;
    }
  }
  __syncthreads();
  for (int r = wv * 2; r < wv * 2 + 2; ++r) {
    float ssum = 0.f;
    for (int t = lane; t < SEQ; t += 32) ssum += sbuf[r * SEQ + t];
#pragma unroll
    for (int off = 16; off >= 1; off >>= 1) ssum += __shfl_xor(ssum, off, 32);
    float inv = 1.f / ssum;
    const int d0 = lane * 8;
    float cx[8];
#pragma unroll
    for (int i = 0; i < 8; ++i) cx[i] = 0.f;
    for (int t = 0; t < SEQ; ++t) {
      float wgt = sbuf[r * SEQ + t] * inv;
      const _Float16* hp = hs + ((size_t)(b0 + r) * SEQ + t) * HID + d0;
#pragma unroll
      for (int i = 0; i < 8; ++i) cx[i] += wgt * (float)hp[i];
    }
#pragma unroll
    for (int i = 0; i < 8; ++i) ctx[(size_t)(b0 + r) * HID + i + d0] = cx[i];
  }
}

// --------------------------- per-batch: norm + MHA + MLP + pool + final ----
__global__ __launch_bounds__(256) void final_kernel(
    const float* __restrict__ ctxs, const float* __restrict__ ctxm,
    const float* __restrict__ nw, const float* __restrict__ nb,
    const float* __restrict__ mwt,
    const _Float16* __restrict__ ipw, const float* __restrict__ ipb,
    const _Float16* __restrict__ opw, const float* __restrict__ opb,
    const _Float16* __restrict__ w1, const float* __restrict__ b1,
    const _Float16* __restrict__ w2, const float* __restrict__ b2,
    const float* __restrict__ fw, const float* __restrict__ fb,
    float* __restrict__ outp) {
  __shared__ __align__(16) char smem[64 * 1024];
  _Float16* mlh = (_Float16*)smem;             // 32x256 ml (f16)
  _Float16* P0  = (_Float16*)(smem + 16384);   // q, later attn
  _Float16* P1  = (_Float16*)(smem + 32768);   // k, later h1
  _Float16* P2  = (_Float16*)(smem + 49152);   // v, later mlp hidden chunk
  float* pooled = (float*)smem;                // reuse ml region once dead
  float* red    = (float*)(smem + 16384);      // scratch (pre-qkv)
  float* red2   = (float*)(smem + 49152);      // scratch (post-mlp)
  const int tid = threadIdx.x, lane = tid & 31, wv = tid >> 5, hi = lane >> 4;
  const int b = blockIdx.x;
  const float* cvec = ctxs + (size_t)b * HID;
  const float* mvec = ctxm + (size_t)b * HID;

  // mean / unbiased std over 256 elements
  float v = cvec[tid];
  float s1 = v, s2 = v * v;
#pragma unroll
  for (int off = 16; off >= 1; off >>= 1) {
    s1 += __shfl_xor(s1, off, 32);
    s2 += __shfl_xor(s2, off, 32);
  }
  if (lane == 0) { red[wv] = s1; red[8 + wv] = s2; }
  __syncthreads();
  float sum = 0.f, sq = 0.f;
  for (int i = 0; i < 8; ++i) { sum += red[i]; sq += red[8 + i]; }
  float mean = sum * (1.f / 256.f);
  float var  = (sq - sum * mean) * (1.f / 255.f);
  float istd = 1.f / (sqrtf(fmaxf(var, 0.f)) + 1e-8f);
  float mw = mwt[0];

  for (int idx = tid; idx < NSTK * HID; idx += 256) {
    int h = idx & 255;
    float ml = nw[idx] * ((cvec[h] - mean) * istd) + nb[idx] + mw * mvec[h];
    mlh[idx] = (_Float16)ml;
  }
  __syncthreads();

  // qkv = ml @ in_proj_w.T + b  (2 M-tiles x 48 N-tiles)
  for (int i = 0; i < 12; ++i) {
    int tt = wv + 8 * i;
    int m0 = (tt & 1) * 16, ng0 = (tt >> 1) * 16;
    v8f acc = zero8();
#pragma unroll
    for (int kk = 0; kk < 8; ++kk) {
      v16h a  = load_frag_h(mlh, HID, m0, kk * 32);
      v16h bb = load_frag_h(ipw, HID, ng0, kk * 32);
      acc = wmma16(a, bb, acc);
    }
    int n = ng0 + (lane & 15);
    float bv = ipb[n];
    _Float16* plane = (n < 256) ? P0 : (n < 512 ? P1 : P2);
    int pc = n & 255;
#pragma unroll
    for (int vv = 0; vv < 8; ++vv)
      plane[(m0 + vv + 8 * hi) * HID + pc] = (_Float16)(acc[vv] + bv);
  }
  __syncthreads();

  // attention: wave = head, lane = query row; 32x32 scores in registers
  {
    const int hc0 = wv * DH;
    const int r = lane;
    float q[DH];
#pragma unroll
    for (int d = 0; d < DH; ++d) q[d] = (float)P0[r * HID + hc0 + d];
    float sc[NSTK];
    const float scl = 0.17677669529663687f;  // 1/sqrt(32)
    for (int kr = 0; kr < NSTK; ++kr) {
      float s = 0.f;
#pragma unroll
      for (int d = 0; d < DH; ++d) s += q[d] * (float)P1[kr * HID + hc0 + d];
      sc[kr] = s * scl;
    }
    float mx = sc[0];
#pragma unroll
    for (int kr = 1; kr < NSTK; ++kr) mx = fmaxf(mx, sc[kr]);
    float se = 0.f;
#pragma unroll
    for (int kr = 0; kr < NSTK; ++kr) { sc[kr] = __expf(sc[kr] - mx); se += sc[kr]; }
    float inv = 1.f / se;
    float av[DH];
#pragma unroll
    for (int d = 0; d < DH; ++d) av[d] = 0.f;
    for (int kr = 0; kr < NSTK; ++kr) {
      float wk = sc[kr] * inv;
#pragma unroll
      for (int d = 0; d < DH; ++d) av[d] += wk * (float)P2[kr * HID + hc0 + d];
    }
#pragma unroll
    for (int d = 0; d < DH; ++d) P0[r * HID + hc0 + d] = (_Float16)av[d];
  }
  __syncthreads();

  // att_out = attn @ out_proj_w.T + b; h1 = ml + att_out -> P1 (f16)
  for (int i = 0; i < 4; ++i) {
    int tt = wv + 8 * i;
    int m0 = (tt & 1) * 16, n0 = (tt >> 1) * 16;
    v8f acc = zero8();
#pragma unroll
    for (int kk = 0; kk < 8; ++kk) {
      v16h a  = load_frag_h(P0, HID, m0, kk * 32);
      v16h bb = load_frag_h(opw, HID, n0, kk * 32);
      acc = wmma16(a, bb, acc);
    }
    int n = n0 + (lane & 15);
    float bv = opb[n];
#pragma unroll
    for (int vv = 0; vv < 8; ++vv) {
      int m = m0 + vv + 8 * hi;
      float h1 = acc[vv] + bv + (float)mlh[m * HID + n];
      P1[m * HID + n] = (_Float16)h1;
    }
  }
  __syncthreads();
  if (tid < 256) pooled[tid] = 0.f;   // ml region dead now
  __syncthreads();

  // MLP, K-chunked: hidden chunk (32x256) in P2, partial acc in registers
  v8f acc2[4];
#pragma unroll
  for (int i = 0; i < 4; ++i) acc2[i] = zero8();
  for (int c = 0; c < 4; ++c) {
    for (int i = 0; i < 4; ++i) {
      int tt = wv + 8 * i;
      int m0 = (tt & 1) * 16, n0 = (tt >> 1) * 16;
      v8f acc = zero8();
#pragma unroll
      for (int kk = 0; kk < 8; ++kk) {
        v16h a  = load_frag_h(P1, HID, m0, kk * 32);
        v16h bb = load_frag_h(w1, HID, c * 256 + n0, kk * 32);
        acc = wmma16(a, bb, acc);
      }
      int n = n0 + (lane & 15);
      float bv = b1[c * 256 + n];
#pragma unroll
      for (int vv = 0; vv < 8; ++vv)
        P2[(m0 + vv + 8 * hi) * HID + n] = (_Float16)fmaxf(acc[vv] + bv, 0.f);
    }
    __syncthreads();
    for (int i = 0; i < 4; ++i) {
      int tt = wv + 8 * i;
      int m0 = (tt & 1) * 16, n0 = (tt >> 1) * 16;
      v8f acc = acc2[i];
#pragma unroll
      for (int kk = 0; kk < 8; ++kk) {
        v16h a  = load_frag_h(P2, HID, m0, kk * 32);
        v16h bb = load_frag_h(w2, GATE4, n0, c * 256 + kk * 32);
        acc = wmma16(a, bb, acc);
      }
      acc2[i] = acc;
    }
    __syncthreads();
  }

  // out = tanh(h1 + mlp_out); pool over 32 tokens
  for (int i = 0; i < 4; ++i) {
    int tt = wv + 8 * i;
    int m0 = (tt & 1) * 16, n0 = (tt >> 1) * 16;
    int n = n0 + (lane & 15);
    float bv = b2[n];
    float psum = 0.f;
#pragma unroll
    for (int vv = 0; vv < 8; ++vv) {
      int m = m0 + vv + 8 * hi;
      psum += tanhf(acc2[i][vv] + bv + (float)P1[m * HID + n]);
    }
    atomicAdd(&pooled[n], psum);
  }
  __syncthreads();
  float val = pooled[tid] * (1.f / 32.f) * fw[tid];
#pragma unroll
  for (int off = 16; off >= 1; off >>= 1) val += __shfl_xor(val, off, 32);
  if (lane == 0) red2[wv] = val;
  __syncthreads();
  if (tid == 0) {
    float tot = 0.f;
    for (int i = 0; i < 8; ++i) tot += red2[i];
    outp[b] = tot + fb[0];
  }
}

// ------------------------------------------------------------------ launch ----
extern "C" void kernel_launch(void* const* d_in, const int* in_sizes, int n_in,
                              void* d_out, int out_size, void* d_ws, size_t ws_size,
                              hipStream_t stream) {
  (void)in_sizes; (void)n_in; (void)out_size; (void)ws_size;
  const float* x    = (const float*)d_in[0];
  char* ws = (char*)d_ws;
  size_t off = 0;
  auto alloc = [&](size_t bytes) { size_t o = off; off += (bytes + 255) & ~(size_t)255; return o; };

  _Float16* stock_tr_h = (_Float16*)(ws + alloc((size_t)DIN * DIN * 2));
  _Float16* macro_tr_h = (_Float16*)(ws + alloc((size_t)DIN * DIN * 2));
  _Float16* s_wih_h    = (_Float16*)(ws + alloc((size_t)GATE4 * DIN * 2));
  _Float16* s_whh_h    = (_Float16*)(ws + alloc((size_t)GATE4 * HID * 2));
  _Float16* m_wih_h    = (_Float16*)(ws + alloc((size_t)GATE4 * DIN * 2));
  _Float16* m_whh_h    = (_Float16*)(ws + alloc((size_t)GATE4 * HID * 2));
  _Float16* ip_h       = (_Float16*)(ws + alloc((size_t)3 * HID * HID * 2));
  _Float16* op_h       = (_Float16*)(ws + alloc((size_t)HID * HID * 2));
  _Float16* w1_h       = (_Float16*)(ws + alloc((size_t)GATE4 * HID * 2));
  _Float16* w2_h       = (_Float16*)(ws + alloc((size_t)HID * GATE4 * 2));
  float*    sb4        = (float*)(ws + alloc((size_t)GATE4 * 4));
  float*    mb4        = (float*)(ws + alloc((size_t)GATE4 * 4));
  float*    ctxs       = (float*)(ws + alloc((size_t)BATCH * HID * 4));
  float*    ctxm       = (float*)(ws + alloc((size_t)BATCH * HID * 4));
  _Float16* act_h      = (_Float16*)(ws + alloc((size_t)BATCH * SEQ * DIN * 2));
  _Float16* gx         = (_Float16*)(ws + alloc((size_t)BATCH * SEQ * GATE4 * 2));
  _Float16* hs         = (_Float16*)(ws + alloc((size_t)BATCH * SEQ * HID * 2));

  auto cvt = [&](int idx, _Float16* dst, int n) {
    cvt_f32_f16<<<(n + 255) / 256, 256, 0, stream>>>((const float*)d_in[idx], dst, n);
  };
  cvt(1, stock_tr_h, DIN * DIN);
  cvt(3, macro_tr_h, DIN * DIN);
  cvt(5, s_wih_h, GATE4 * DIN);
  cvt(6, s_whh_h, GATE4 * HID);
  cvt(9, m_wih_h, GATE4 * DIN);
  cvt(10, m_whh_h, GATE4 * HID);
  cvt(16, ip_h, 3 * HID * HID);
  cvt(18, op_h, HID * HID);
  cvt(20, w1_h, GATE4 * HID);
  cvt(22, w2_h, HID * GATE4);
  bias_comb<<<(GATE4 + 255) / 256, 256, 0, stream>>>(
      (const float*)d_in[7], (const float*)d_in[8], sb4, GATE4);
  bias_comb<<<(GATE4 + 255) / 256, 256, 0, stream>>>(
      (const float*)d_in[11], (const float*)d_in[12], mb4, GATE4);

  const int mtiles = (BATCH * SEQ) / 16;  // 8192

  // ---- stock phase ----
  transform_kernel<<<mtiles, 256, 0, stream>>>(x, stock_tr_h, (const float*)d_in[2], act_h);
  gates_kernel<<<mtiles, 256, 0, stream>>>(act_h, s_wih_h, gx);
  lstm_kernel<<<BATCH / 16, 256, 0, stream>>>(gx, s_whh_h, sb4, hs, ctxs);

  // ---- macro phase (reuses act/gx/hs buffers; stream-ordered) ----
  transform_kernel<<<mtiles, 256, 0, stream>>>(x, macro_tr_h, (const float*)d_in[4], act_h);
  gates_kernel<<<mtiles, 256, 0, stream>>>(act_h, m_wih_h, gx);
  lstm_kernel<<<BATCH / 16, 256, 0, stream>>>(gx, m_whh_h, mb4, hs, ctxm);

  // ---- norm + MHA + MLP + pool + final ----
  final_kernel<<<BATCH, 256, 0, stream>>>(
      ctxs, ctxm, (const float*)d_in[13], (const float*)d_in[14],
      (const float*)d_in[15], ip_h, (const float*)d_in[17], op_h,
      (const float*)d_in[19], w1_h, (const float*)d_in[21], w2_h,
      (const float*)d_in[23], (const float*)d_in[24], (const float*)d_in[25],
      (float*)d_out);
}
